// GCNLayer_4217657884682
// MI455X (gfx1250) — matrix operand order repaired and checked by reading, not compiled, not run on hardware
//
#include <hip/hip_runtime.h>

#define N_NODES 100000
#define N_EDGES 1600000
#define D 128
#define N_WAVES_M 6250   // N_NODES / 16

typedef __bf16 v16bf __attribute__((ext_vector_type(16)));
typedef __bf16 v2bf  __attribute__((ext_vector_type(2)));
typedef float  v8f   __attribute__((ext_vector_type(8)));

union FragB16 { v16bf v; unsigned int u[8]; };

// Pack two f32 into a dword of two bf16 (RNE). Use HW packed convert when the
// toolchain exposes it for gfx1250; otherwise manual round-to-nearest-even.
__device__ __forceinline__ unsigned int pack_bf16(float a, float b) {
#if __has_builtin(__builtin_amdgcn_cvt_pk_bf16_f32)
    v2bf p = __builtin_amdgcn_cvt_pk_bf16_f32(a, b);
    return __builtin_bit_cast(unsigned int, p);
#else
    unsigned int ua = __builtin_bit_cast(unsigned int, a);
    ua += 0x7FFFu + ((ua >> 16) & 1u);
    unsigned int ub = __builtin_bit_cast(unsigned int, b);
    ub += 0x7FFFu + ((ub >> 16) & 1u);
    return (ua >> 16) | (ub & 0xFFFF0000u);
#endif
}

// ---------------- degree / norm ----------------
__global__ void k_deg_init(float* deg) {
    int i = blockIdx.x * blockDim.x + threadIdx.x;
    if (i < N_NODES) deg[i] = 1.0f;   // self-loop contributes 1
}

__global__ void k_deg_accum(const int* __restrict__ col, float* deg) {
    int e = blockIdx.x * blockDim.x + threadIdx.x;
    if (e < N_EDGES) atomicAdd(&deg[col[e]], 1.0f);
}

__global__ void k_dinv(float* deg) {
    int i = blockIdx.x * blockDim.x + threadIdx.x;
    if (i < N_NODES) deg[i] = __frsqrt_rn(deg[i]);  // deg >= 1 always
}

// ---------------- W pre-swizzle into B-fragment layout ----------------
// Layout: [nt(8)][kc(4)][lane(32)][slot(8)] of packed bf16 pairs.
// B[k][n] = W[n][k] (h = x @ W^T). Lanes 0-15: N=lane, K=0..7,16..23; lanes 16-31: K=8..15,24..31
__global__ void k_wswiz(const float* __restrict__ W, unsigned int* __restrict__ wsw) {
    int idx = blockIdx.x * blockDim.x + threadIdx.x;
    if (idx >= 8 * 4 * 32 * 8) return;
    int v2   = idx & 7;
    int lane = (idx >> 3) & 31;
    int kc   = (idx >> 8) & 3;
    int nt   = idx >> 10;
    int n = nt * 16 + (lane & 15);
    int g = lane >> 4;
    int k = kc * 32 + ((v2 < 4) ? (8 * g + 2 * v2) : (16 + 8 * g + 2 * (v2 - 4)));
    wsw[idx] = pack_bf16(W[n * D + k], W[n * D + k + 1]);
}

// ---------------- WMMA GEMM: h = x @ W^T ----------------
// One wave computes 16 rows x 128 cols. 8 waves / block. No LDS needed:
// A fragments built straight from global x (each wave fully consumes 128B
// segments of 16 rows), B fragments read from the pre-swizzled 32KB table.
__global__ __launch_bounds__(256) void k_gemm(const float* __restrict__ x,
                                              const unsigned int* __restrict__ wsw,
                                              float* __restrict__ h) {
    int wave = blockIdx.x * 8 + (threadIdx.x >> 5);
    if (wave >= N_WAVES_M) return;            // uniform per-wave exit (keeps EXEC full)
    int lane = threadIdx.x & 31;
    int l16  = lane & 15;
    int g    = lane >> 4;

    int m = wave * 16 + l16;                  // global row this lane's A slots cover
    const float* xrow = x + (size_t)m * D;

    // Build 4 A fragments (K chunks of 32), per ISA 16-bit 16x32 A layout.
    v16bf afrag[4];
    #pragma unroll
    for (int kc = 0; kc < 4; ++kc) {
        FragB16 fa;
        #pragma unroll
        for (int v2 = 0; v2 < 8; ++v2) {
            int k = kc * 32 + ((v2 < 4) ? (8 * g + 2 * v2)
                                        : (16 + 8 * g + 2 * (v2 - 4)));
            float2 p = *(const float2*)(xrow + k);
            fa.u[v2] = pack_bf16(p.x, p.y);
        }
        afrag[kc] = fa.v;
    }

    v8f acc[8];
    #pragma unroll
    for (int nt = 0; nt < 8; ++nt) acc[nt] = (v8f){0,0,0,0,0,0,0,0};

    #pragma unroll
    for (int nt = 0; nt < 8; ++nt) {
        #pragma unroll
        for (int kc = 0; kc < 4; ++kc) {
            FragB16 fb;
            const unsigned int* src = wsw + (((nt * 4 + kc) * 32) + lane) * 8;
            *(uint4*)(&fb.u[0]) = *(const uint4*)(src);
            *(uint4*)(&fb.u[4]) = *(const uint4*)(src + 4);
            acc[nt] = __builtin_amdgcn_wmma_f32_16x16x32_bf16(
                false, afrag[kc], false, fb.v, (short)0, acc[nt], false, false);
        }
    }

    // Store: VGPR r -> row m0 + r + 8g, col nt*16 + l16.
    int m0 = wave * 16;
    #pragma unroll
    for (int nt = 0; nt < 8; ++nt) {
        int n = nt * 16 + l16;
        #pragma unroll
        for (int r = 0; r < 8; ++r) {
            h[(size_t)(m0 + r + 8 * g) * D + n] = acc[nt][r];
        }
    }
}

// ---------------- out = dinv^2 * h + b (self-loop + bias) ----------------
__global__ void k_selfinit(const float* __restrict__ h, const float* __restrict__ dinv,
                           const float* __restrict__ b, float* __restrict__ out) {
    int idx = blockIdx.x * blockDim.x + threadIdx.x;   // over N*D/4 float4s
    if (idx >= N_NODES * (D / 4)) return;
    int node = idx >> 5;          // D/4 == 32
    int c4   = idx & 31;
    float s = dinv[node]; s = s * s;
    float4 hv = ((const float4*)h)[idx];
    float4 bv = ((const float4*)b)[c4];
    float4 o;
    o.x = s * hv.x + bv.x;  o.y = s * hv.y + bv.y;
    o.z = s * hv.z + bv.z;  o.w = s * hv.w + bv.w;
    ((float4*)out)[idx] = o;
}

// ---------------- edge scatter: one wave per edge ----------------
__global__ __launch_bounds__(256) void k_scatter(const int* __restrict__ row,
                                                 const int* __restrict__ col,
                                                 const float* __restrict__ dinv,
                                                 const float* __restrict__ h,
                                                 float* __restrict__ out) {
    int e    = (blockIdx.x * 256 + threadIdx.x) >> 5;
    int lane = threadIdx.x & 31;
    if (e >= N_EDGES) return;
    // e is wave-uniform: force it into an SGPR so row/col/dinv become scalar loads.
    int eu = __builtin_amdgcn_readfirstlane(e);
    int r = row[eu];
    int c = col[eu];
    float norm = dinv[r] * dinv[c];
    float4 m = ((const float4*)(h + (size_t)r * D))[lane];   // coalesced 512B/wave
    m.x *= norm; m.y *= norm; m.z *= norm; m.w *= norm;
    float* op = out + (size_t)c * D + lane * 4;
    atomicAdd(op + 0, m.x);
    atomicAdd(op + 1, m.y);
    atomicAdd(op + 2, m.z);
    atomicAdd(op + 3, m.w);
}

// ---------------- ReLU ----------------
__global__ void k_relu(float* out) {
    int idx = blockIdx.x * blockDim.x + threadIdx.x;
    if (idx >= N_NODES * (D / 4)) return;
    float4 v = ((float4*)out)[idx];
    v.x = fmaxf(v.x, 0.f); v.y = fmaxf(v.y, 0.f);
    v.z = fmaxf(v.z, 0.f); v.w = fmaxf(v.w, 0.f);
    ((float4*)out)[idx] = v;
}

extern "C" void kernel_launch(void* const* d_in, const int* in_sizes, int n_in,
                              void* d_out, int out_size, void* d_ws, size_t ws_size,
                              hipStream_t stream) {
    const float* x    = (const float*)d_in[0];
    const int*   edge = (const int*)d_in[1];
    const float* W    = (const float*)d_in[2];
    const float* b    = (const float*)d_in[3];
    float* out = (float*)d_out;

    const int* row = edge;            // edge_index[0] = sources
    const int* col = edge + N_EDGES;  // edge_index[1] = targets

    char* wsp = (char*)d_ws;
    float*        h   = (float*)wsp;                                   // N*D f32 (51.2 MB)
    float*        deg = (float*)(wsp + (size_t)N_NODES * D * 4);       // N f32 (-> dinv)
    unsigned int* wsw = (unsigned int*)(wsp + (size_t)N_NODES * D * 4
                                            + (size_t)N_NODES * 4);    // 32 KB swizzled W

    int tot4 = N_NODES * (D / 4);

    k_deg_init <<<(N_NODES + 255) / 256, 256, 0, stream>>>(deg);
    k_deg_accum<<<(N_EDGES + 255) / 256, 256, 0, stream>>>(col, deg);
    k_dinv     <<<(N_NODES + 255) / 256, 256, 0, stream>>>(deg);
    k_wswiz    <<<(8192 + 255) / 256, 256, 0, stream>>>(W, wsw);
    k_gemm     <<<(N_WAVES_M + 7) / 8, 256, 0, stream>>>(x, wsw, h);
    k_selfinit <<<(tot4 + 255) / 256, 256, 0, stream>>>(h, deg, b, out);
    k_scatter  <<<(N_EDGES * 32 + 255) / 256, 256, 0, stream>>>(row, col, deg, h, out);
    k_relu     <<<(tot4 + 255) / 256, 256, 0, stream>>>(out);
}
